// NonStationaryTransformer_82411832476319
// MI455X (gfx1250) — matrix operand order repaired and checked by reading, not compiled, not run on hardware
//
#include <hip/hip_runtime.h>
#include <hip/hip_bf16.h>

// ---------------- model constants ----------------
#define BB   16
#define SEQL 1024      // SEQ and LABEL+PRED are both 1024
#define EINC 128
#define DM   512
#define NH   8
#define EH   64
#define DFFN 2048
#define PH   128
#define MTOT (BB * SEQL)   // 16384 rows for enc and dec streams
#define FFCH 4096          // FFN row chunk

typedef __bf16 bf16;
typedef bf16 v16bf __attribute__((ext_vector_type(16)));
typedef float v8f  __attribute__((ext_vector_type(8)));
typedef unsigned int u32x4 __attribute__((ext_vector_type(4)));

union FragU { v16bf v; u32x4 q[2]; };

static __device__ __forceinline__ v8f v8f_zero() {
    v8f z = {0.f,0.f,0.f,0.f,0.f,0.f,0.f,0.f};
    return z;
}

// A-operand fragment (16x32 bf16): lane group g holds k in [8g,8g+8) and [16+8g,16+8g+8)
static __device__ __forceinline__ v16bf fragA(const bf16* rowbase, int g) {
    FragU f;
    f.q[0] = *(const u32x4*)(rowbase + 8*g);
    f.q[1] = *(const u32x4*)(rowbase + 16 + 8*g);
    return f.v;
}
// B-operand fragment (32x16 bf16): lane group g holds k in [8g,8g+8) and [16+8g,16+8g+8)
static __device__ __forceinline__ v16bf fragB(const bf16* rowbase, int g) {
    FragU f;
    f.q[0] = *(const u32x4*)(rowbase + 8*g);
    f.q[1] = *(const u32x4*)(rowbase + 16 + 8*g);
    return f.v;
}

static __device__ __forceinline__ v8f wmma_bf16(v16bf a, v16bf b, v8f c) {
    return __builtin_amdgcn_wmma_f32_16x16x32_bf16(false, a, false, b, (short)0, c, false, false);
}

// CDNA5 async DMA: copy 16B/lane global -> LDS, tracked by ASYNCcnt
static __device__ __forceinline__ void async_b128(unsigned lds_addr, const bf16* gp) {
    asm volatile("global_load_async_to_lds_b128 %0, %1, off"
                 :: "v"(lds_addr), "v"((unsigned long long)(size_t)gp) : "memory");
}

static __device__ __forceinline__ float rmax16(float a) {
    a = fmaxf(a, __shfl_xor(a, 1, 32));
    a = fmaxf(a, __shfl_xor(a, 2, 32));
    a = fmaxf(a, __shfl_xor(a, 4, 32));
    a = fmaxf(a, __shfl_xor(a, 8, 32));
    return a;
}
static __device__ __forceinline__ float rsum16(float a) {
    a += __shfl_xor(a, 1, 32);
    a += __shfl_xor(a, 2, 32);
    a += __shfl_xor(a, 4, 32);
    a += __shfl_xor(a, 8, 32);
    return a;
}

// weight convert: W f32 [K,N] -> Wt bf16 [N,K] (K power of two)
__global__ void wt_kernel(const float* __restrict__ W, bf16* __restrict__ Wt,
                          int kshift, int N)
{
    long o = (long)blockIdx.x * 256 + threadIdx.x;   // o = n*K + k
    int  k = (int)(o & ((1 << kshift) - 1));
    long n = o >> kshift;
    Wt[o] = (bf16)W[(long)k * N + n];
}

// ---------------- WMMA GEMM with async double-buffered LDS staging ----------------
// C[M,N] = act(A[M,K]@W + bias + resid); A bf16 [M,K], Wt bf16 [N,K] (pre-transposed)
// grid(M/128, N/128), block 256 (8 waves). K%32==0, N%128==0, M%128==0.
__global__ __launch_bounds__(256) void gemm_bf16_kernel(
    const bf16* __restrict__ A, const bf16* __restrict__ Wt,
    const float* __restrict__ bias, const float* __restrict__ resid,
    float* __restrict__ Cf, bf16* __restrict__ Cb, int N, int K, int act)
{
    __shared__ __align__(16) bf16 la[2][128 * 40];
    __shared__ __align__(16) bf16 lw[2][128 * 40];
    const int tid = threadIdx.x;
    const int wv = tid >> 5, lane = tid & 31;
    const int g = lane >> 4, m = lane & 15;
    const int wr = (wv >> 1) * 32;
    const int wc = (wv & 1) * 64;
    const long rowBase = (long)blockIdx.x * 128;
    const long colBase = (long)blockIdx.y * 128;

    // chunk geometry shared by A and W staging: 512 chunks of 8 halves, 2 per thread
    const int c0 = (tid & 3) * 8, r0 = tid >> 2;           // chunk tid
    const int c1 = ((tid + 256) & 3) * 8, r1 = (tid + 256) >> 2;

    auto stage = [&](int kk, int buf) {
        async_b128((unsigned)(size_t)&la[buf][r0 * 40 + c0], A  + (rowBase + r0) * (long)K + kk + c0);
        async_b128((unsigned)(size_t)&la[buf][r1 * 40 + c1], A  + (rowBase + r1) * (long)K + kk + c1);
        async_b128((unsigned)(size_t)&lw[buf][r0 * 40 + c0], Wt + (colBase + r0) * (long)K + kk + c0);
        async_b128((unsigned)(size_t)&lw[buf][r1 * 40 + c1], Wt + (colBase + r1) * (long)K + kk + c1);
    };

    v8f acc[2][4];
#pragma unroll
    for (int t = 0; t < 2; t++)
#pragma unroll
        for (int c = 0; c < 4; c++) acc[t][c] = v8f_zero();

    const int nt = K >> 5;
    stage(0, 0);
    for (int i = 0; i < nt; i++) {
        const int cur = i & 1;
        if (i + 1 < nt) {
            stage((i + 1) << 5, cur ^ 1);
            asm volatile("s_wait_asynccnt 0x4" ::: "memory");
        } else {
            asm volatile("s_wait_asynccnt 0x0" ::: "memory");
        }
        __syncthreads();   // tile i visible to all waves

        v16bf af[2], bfr[4];
#pragma unroll
        for (int t = 0; t < 2; t++) af[t] = fragA(&la[cur][(wr + 16 * t + m) * 40], g);
#pragma unroll
        for (int c = 0; c < 4; c++) bfr[c] = fragB(&lw[cur][(wc + 16 * c + m) * 40], g);
#pragma unroll
        for (int t = 0; t < 2; t++)
#pragma unroll
            for (int c = 0; c < 4; c++)
                acc[t][c] = wmma_bf16(af[t], bfr[c], acc[t][c]);
        __syncthreads();   // all reads of buf done before it is overwritten
    }

#pragma unroll
    for (int t = 0; t < 2; t++)
#pragma unroll
        for (int c = 0; c < 4; c++) {
            const long col = colBase + wc + 16 * c + m;
            const float bc = bias ? bias[col] : 0.f;
#pragma unroll
            for (int r = 0; r < 8; r++) {
                const long row = rowBase + wr + 16 * t + g * 8 + r;
                float v = acc[t][c][r] + bc;
                if (resid) v += resid[row * N + col];
                if (act == 1)      v = fmaxf(v, 0.f);
                else if (act == 2) v = 0.5f * v * (1.f + erff(v * 0.70710678118654752f));
                if (Cf) Cf[row * N + col] = v;
                if (Cb) Cb[row * N + col] = (bf16)v;
            }
        }
}

// ---------------- Flash attention (de-stationary), bf16 in/out ----------------
// Q,K: bf16 [B,1024,512]; Vt: bf16 [B*H, 64, 1024]; O: bf16 [B,1024,512]
__global__ __launch_bounds__(128) void attn_kernel(
    const bf16* __restrict__ Q, const bf16* __restrict__ Km,
    const bf16* __restrict__ Vt, const float* __restrict__ tau,
    const float* __restrict__ delta, bf16* __restrict__ O)
{
    __shared__ __align__(16) bf16 plds[4 * 16 * 40];
    const int wv = threadIdx.x >> 5, lane = threadIdx.x & 31;
    const int g = lane >> 4, m = lane & 15;
    const int b = blockIdx.x >> 3, h = blockIdx.x & 7;
    const int qrow = blockIdx.y * 64 + wv * 16;

    const bf16* qp = Q + ((long)(b * SEQL + qrow + m)) * DM + h * EH;
    v16bf qa0 = fragA(qp, g);
    v16bf qa1 = fragA(qp + 32, g);
    const float taus = tau[b] * 0.125f;

    float tmax[8], lsum[8];
    v8f o[4];
#pragma unroll
    for (int r = 0; r < 8; r++) { tmax[r] = -1e30f; lsum[r] = 0.f; }
#pragma unroll
    for (int c = 0; c < 4; c++) o[c] = v8f_zero();
    bf16* pw = &plds[wv * 640];

    for (int kt = 0; kt < SEQL; kt += 32) {
        const bf16* kp0 = Km + ((long)(b * SEQL + kt + m)) * DM + h * EH;
        const bf16* kp1 = kp0 + 16 * DM;
        v8f s0 = v8f_zero(), s1 = v8f_zero();
        s0 = wmma_bf16(qa0, fragB(kp0, g), s0);
        s0 = wmma_bf16(qa1, fragB(kp0 + 32, g), s0);
        s1 = wmma_bf16(qa0, fragB(kp1, g), s1);
        s1 = wmma_bf16(qa1, fragB(kp1 + 32, g), s1);

        float d0 = 0.f, d1 = 0.f;
        if (delta) {
            d0 = delta[b * SEQL + kt + m] * 0.125f;
            d1 = delta[b * SEQL + kt + 16 + m] * 0.125f;
        }
#pragma unroll
        for (int r = 0; r < 8; r++) {
            float a0 = s0[r] * taus + d0;
            float a1 = s1[r] * taus + d1;
            float mx = rmax16(fmaxf(a0, a1));
            float mn = fmaxf(tmax[r], mx);
            float corr = __expf(tmax[r] - mn);
            tmax[r] = mn;
            float p0 = __expf(a0 - mn), p1 = __expf(a1 - mn);
            float rs = rsum16(p0 + p1);
            lsum[r] = lsum[r] * corr + rs;
#pragma unroll
            for (int c = 0; c < 4; c++) o[c][r] *= corr;
            pw[(g * 8 + r) * 40 + m]      = (bf16)p0;
            pw[(g * 8 + r) * 40 + 16 + m] = (bf16)p1;
        }
        asm volatile("s_wait_dscnt 0x0" ::: "memory");
        v16bf pf = fragA(&pw[m * 40], g);
        const bf16* vb = Vt + ((long)blockIdx.x * 64 + m) * SEQL + kt;
#pragma unroll
        for (int c = 0; c < 4; c++)
            o[c] = wmma_bf16(pf, fragB(vb + (long)c * 16 * SEQL, g), o[c]);
    }
#pragma unroll
    for (int c = 0; c < 4; c++)
#pragma unroll
        for (int r = 0; r < 8; r++) {
            long row = qrow + g * 8 + r;
            long col = h * EH + c * 16 + m;
            O[((long)b * SEQL + row) * DM + col] = (bf16)(o[c][r] / lsum[r]);
        }
}

// ---------------- LayerNorm (dim 512), dual f32 + bf16 output ----------------
__global__ __launch_bounds__(256) void ln_kernel(
    const float* __restrict__ X, const float* __restrict__ gam,
    const float* __restrict__ bet, float* __restrict__ Y, bf16* __restrict__ Yb)
{
    const int row = blockIdx.x * 8 + (threadIdx.x >> 5);
    const int lane = threadIdx.x & 31;
    const float* x = X + (long)row * DM;
    float s = 0.f, q = 0.f;
#pragma unroll
    for (int i = 0; i < 16; i++) { float v = x[lane + i * 32]; s += v; q += v * v; }
#pragma unroll
    for (int msk = 16; msk >= 1; msk >>= 1) {
        s += __shfl_xor(s, msk, 32);
        q += __shfl_xor(q, msk, 32);
    }
    float mean = s * (1.f / DM);
    float var  = q * (1.f / DM) - mean * mean;
    float rstd = rsqrtf(var + 1e-5f);
#pragma unroll
    for (int i = 0; i < 16; i++) {
        int d = lane + i * 32;
        float v = (x[d] - mean) * rstd * gam[d] + bet[d];
        if (Y)  Y[(long)row * DM + d] = v;
        if (Yb) Yb[(long)row * DM + d] = (bf16)v;
    }
}

// ---------------- series stats + normalized bf16 input ----------------
__global__ void stats_kernel(const float* __restrict__ x, bf16* __restrict__ xeb,
                             float* __restrict__ meanb, float* __restrict__ stdb)
{
    const int b = blockIdx.x, e = threadIdx.x;
    float s = 0.f, q = 0.f;
    for (int t = 0; t < SEQL; t++) {
        float v = x[((long)b * SEQL + t) * EINC + e];
        s += v; q += v * v;
    }
    float m = s * (1.f / SEQL);
    float var = q * (1.f / SEQL) - m * m;
    float sd = sqrtf(var + 1e-5f);
    meanb[b * EINC + e] = m;
    stdb[b * EINC + e] = sd;
    float inv = 1.f / sd;
    for (int t = 0; t < SEQL; t++) {
        long idx = ((long)b * SEQL + t) * EINC + e;
        xeb[idx] = (bf16)((x[idx] - m) * inv);
    }
}

// x_dec_new (bf16) = concat(normalize(x_enc)[:, -512:, :], zeros)
__global__ void xdn_kernel(const float* __restrict__ x, const float* __restrict__ meanb,
                           const float* __restrict__ stdb, bf16* __restrict__ xdn)
{
    long idx = (long)blockIdx.x * 256 + threadIdx.x;   // over 16384*128
    int c = (int)(idx & 127);
    long row = idx >> 7;
    int s = (int)(row & 1023), b = (int)(row >> 10);
    float v = 0.f;
    if (s < 512)
        v = (x[((long)b * SEQL + 512 + s) * EINC + c] - meanb[b * EINC + c]) / stdb[b * EINC + c];
    xdn[idx] = (bf16)v;
}

// circular conv over feature axis
__global__ void conv_kernel(const float* __restrict__ x, const float* __restrict__ cw,
                            float* __restrict__ c)
{
    const int b = blockIdx.x, e = threadIdx.x;
    const int em = (e + 127) & 127, ep = (e + 1) & 127;
    float acc = 0.f;
    for (int i = 0; i < SEQL; i++) {
        const float* xr = x + ((long)b * SEQL + i) * EINC;
        const float* w = cw + i * 3;
        acc += xr[em] * w[0] + xr[e] * w[1] + xr[ep] * w[2];
    }
    c[b * EINC + e] = acc;
}

__global__ void mlp0_kernel(const float* __restrict__ c, const float* __restrict__ stat,
                            const float* __restrict__ w0, const float* __restrict__ b0,
                            float* __restrict__ y)
{
    const int b = blockIdx.x, j = threadIdx.x;
    float acc = b0[j];
    for (int p = 0; p < 128; p++) acc += c[b * 128 + p] * w0[p * PH + j];
    for (int p = 0; p < 128; p++) acc += stat[b * 128 + p] * w0[(128 + p) * PH + j];
    y[b * PH + j] = fmaxf(acc, 0.f);
}

__global__ void tau_kernel(const float* __restrict__ y, const float* __restrict__ w1,
                           float* __restrict__ tau)
{
    int b = threadIdx.x;
    if (b < BB) {
        float acc = 0.f;
        for (int p = 0; p < PH; p++) acc += y[b * PH + p] * w1[p];
        tau[b] = __expf(acc);
    }
}

__global__ void delta_kernel(const float* __restrict__ y, const float* __restrict__ w1,
                             float* __restrict__ delta)
{
    int b = blockIdx.y;
    int s = blockIdx.x * 128 + threadIdx.x;
    float acc = 0.f;
    for (int p = 0; p < PH; p++) acc += y[b * PH + p] * w1[p * SEQL + s];
    delta[b * SEQL + s] = acc;
}

// positional encoding + time-mark embedding + biases
__global__ void embextra_kernel(const float* __restrict__ xmark, const float* __restrict__ wt,
                                const float* __restrict__ bt, const float* __restrict__ bv,
                                float* __restrict__ out)
{
    long idx = (long)blockIdx.x * 256 + threadIdx.x;   // over 16384*512
    int d = (int)(idx & 511);
    long row = idx >> 9;
    int s = (int)(row & 1023);
    float kf = (float)(d & ~1);
    float f = __expf(kf * (-9.210340371976184f / 512.0f));
    float ang = (float)s * f;
    float pe = (d & 1) ? __cosf(ang) : __sinf(ang);
    const float* xm = xmark + row * 4;
    float acc = pe + bt[d] + bv[d];
#pragma unroll
    for (int t = 0; t < 4; t++) acc += xm[t] * wt[t * DM + d];
    out[idx] = acc;
}

// V f32 [B,1024,512] -> Vt bf16 [B*H, 64, 1024]
__global__ void vt_kernel(const float* __restrict__ V, bf16* __restrict__ Vt)
{
    long t = (long)blockIdx.x * 256 + threadIdx.x;
    int s = (int)(t & 1023);
    int e = (int)((t >> 10) & 63);
    int bh = (int)(t >> 16);
    int b = bh >> 3, h = bh & 7;
    Vt[t] = (bf16)V[((long)b * SEQL + s) * DM + h * EH + e];
}

__global__ void final_kernel(const float* __restrict__ Z, const float* __restrict__ stdb,
                             const float* __restrict__ meanb, float* __restrict__ out)
{
    long idx = (long)blockIdx.x * 256 + threadIdx.x;   // 16*512*128
    int c = (int)(idx & 127);
    int sp = (int)((idx >> 7) & 511);
    int b = (int)(idx >> 16);
    float v = Z[((long)b * SEQL + 512 + sp) * 128 + c];
    out[idx] = v * stdb[b * EINC] + meanb[b * EINC];
}

// ---------------- host orchestration ----------------
extern "C" void kernel_launch(void* const* d_in, const int* in_sizes, int n_in,
                              void* d_out, int out_size, void* d_ws, size_t ws_size,
                              hipStream_t stream)
{
    (void)in_sizes; (void)n_in; (void)out_size; (void)ws_size;
    auto P = [&](int i) -> const float* { return (const float*)d_in[i]; };
    const float* x_enc      = P(0);
    const float* x_mark_enc = P(1);
    const float* x_mark_dec = P(3);

    char* wsp = (char*)d_ws;
    size_t off = 0;
    auto alloc = [&](size_t bytes) -> void* {
        void* p = wsp + off;
        off += (bytes + 255) & ~(size_t)255;
        return p;
    };
    bf16*  xeb   = (bf16*)alloc((size_t)MTOT * EINC * 2);
    bf16*  xdnb  = (bf16*)alloc((size_t)MTOT * EINC * 2);
    float* meanb = (float*)alloc(BB * EINC * 4);
    float* stdb  = (float*)alloc(BB * EINC * 4);
    float* ctau  = (float*)alloc(BB * EINC * 4);
    float* cdel  = (float*)alloc(BB * EINC * 4);
    float* ytau  = (float*)alloc(BB * PH * 4);
    float* ydel  = (float*)alloc(BB * PH * 4);
    float* tauB  = (float*)alloc(BB * 4);
    float* delB  = (float*)alloc(BB * SEQL * 4);
    float* X     = (float*)alloc((size_t)MTOT * DM * 4);
    float* Y     = (float*)alloc((size_t)MTOT * DM * 4);
    float* Z     = (float*)alloc((size_t)MTOT * DM * 4);
    float* ENC   = (float*)alloc((size_t)MTOT * DM * 4);
    bf16*  Xb    = (bf16*)alloc((size_t)MTOT * DM * 2);
    bf16*  Zb    = (bf16*)alloc((size_t)MTOT * DM * 2);
    bf16*  ENCb  = (bf16*)alloc((size_t)MTOT * DM * 2);
    bf16*  qb    = (bf16*)alloc((size_t)MTOT * DM * 2);
    bf16*  kb    = (bf16*)alloc((size_t)MTOT * DM * 2);
    bf16*  vt    = (bf16*)alloc((size_t)MTOT * DM * 2);
    bf16*  Hb    = (bf16*)alloc((size_t)FFCH * DFFN * 2);

    // ---- one-time weight conversion to transposed bf16 [N,K] ----
    auto cvt = [&](int idx, int K, int N) -> bf16* {
        bf16* w = (bf16*)alloc((size_t)K * N * 2);
        int kshift = __builtin_ctz((unsigned)K);
        wt_kernel<<<(K * N) / 256, 256, 0, stream>>>(P(idx), w, kshift, N);
        return w;
    };
    bf16* WembE = cvt(4, EINC, DM);
    bf16* WembD = cvt(8, EINC, DM);
    bf16 *Wq[2], *Wk[2], *Wv[2], *Wo[2], *W1[2], *W2[2];
    for (int l = 0; l < 2; l++) {
        int pb = 12 + 16 * l;
        Wq[l] = cvt(pb + 0, DM, DM);  Wk[l] = cvt(pb + 2, DM, DM);
        Wv[l] = cvt(pb + 4, DM, DM);  Wo[l] = cvt(pb + 6, DM, DM);
        W1[l] = cvt(pb + 12, DM, DFFN); W2[l] = cvt(pb + 14, DFFN, DM);
    }
    bf16* Sq = cvt(46, DM, DM); bf16* Sk = cvt(48, DM, DM);
    bf16* Sv = cvt(50, DM, DM); bf16* So = cvt(52, DM, DM);
    bf16* Cq = cvt(54, DM, DM); bf16* Ck = cvt(56, DM, DM);
    bf16* Cv = cvt(58, DM, DM); bf16* Co = cvt(60, DM, DM);
    bf16* DW1 = cvt(68, DM, DFFN); bf16* DW2 = cvt(70, DFFN, DM);
    bf16* Wproj = cvt(74, DM, 128);

    auto gemm = [&](const bf16* A, const bf16* Wt, const float* bias, const float* resid,
                    float* Cf, bf16* Cb, int M, int N, int K, int act) {
        gemm_bf16_kernel<<<dim3(M / 128, N / 128), 256, 0, stream>>>(A, Wt, bias, resid, Cf, Cb, N, K, act);
    };
    auto attn = [&](const bf16* q, const bf16* k, const bf16* v, const float* dlt, bf16* out) {
        attn_kernel<<<dim3(BB * NH, SEQL / 64), 128, 0, stream>>>(q, k, v, tauB, dlt, out);
    };
    auto ln = [&](const float* x, int gi, int bi, float* yf, bf16* yb) {
        ln_kernel<<<MTOT / 8, 256, 0, stream>>>(x, P(gi), P(bi), yf, yb);
    };
    auto ffn = [&](const bf16* xb, const float* xf, float* yout,
                   const bf16* w1, int b1i, const bf16* w2, int b2i) {
        for (int ch = 0; ch < MTOT / FFCH; ch++) {
            long ro = (long)ch * FFCH;
            gemm(xb + ro * DM, w1, P(b1i), nullptr, nullptr, Hb, FFCH, DFFN, DM, 2);
            gemm(Hb, w2, P(b2i), xf + ro * DM, yout + ro * DM, nullptr, FFCH, DM, DFFN, 0);
        }
    };

    // 1) series stats + normalized inputs
    stats_kernel<<<BB, EINC, 0, stream>>>(x_enc, xeb, meanb, stdb);
    xdn_kernel<<<(MTOT * EINC) / 256, 256, 0, stream>>>(x_enc, meanb, stdb, xdnb);

    // 2) tau / delta projectors (raw x_enc)
    conv_kernel<<<BB, EINC, 0, stream>>>(x_enc, P(76), ctau);
    conv_kernel<<<BB, EINC, 0, stream>>>(x_enc, P(80), cdel);
    mlp0_kernel<<<BB, PH, 0, stream>>>(ctau, stdb, P(77), P(78), ytau);
    mlp0_kernel<<<BB, PH, 0, stream>>>(cdel, meanb, P(81), P(82), ydel);
    tau_kernel<<<1, 32, 0, stream>>>(ytau, P(79), tauB);
    delta_kernel<<<dim3(SEQL / 128, BB), 128, 0, stream>>>(ydel, P(83), delB);

    // 3) encoder embedding
    embextra_kernel<<<(MTOT * DM) / 256, 256, 0, stream>>>(x_mark_enc, P(6), P(7), P(5), Y);
    gemm(xeb, WembE, nullptr, Y, X, Xb, MTOT, DM, EINC, 0);

    // 4) encoder layers
    for (int l = 0; l < 2; l++) {
        int pb = 12 + 16 * l;
        gemm(Xb, Wq[l], P(pb + 1), nullptr, nullptr, qb, MTOT, DM, DM, 0);
        gemm(Xb, Wk[l], P(pb + 3), nullptr, nullptr, kb, MTOT, DM, DM, 0);
        gemm(Xb, Wv[l], P(pb + 5), nullptr, Y, nullptr, MTOT, DM, DM, 0);
        vt_kernel<<<(MTOT * DM) / 256, 256, 0, stream>>>(Y, vt);
        attn(qb, kb, vt, delB, Zb);
        gemm(Zb, Wo[l], P(pb + 7), X, Y, nullptr, MTOT, DM, DM, 0);
        ln(Y, pb + 8, pb + 9, Z, Zb);
        ffn(Zb, Z, Y, W1[l], pb + 13, W2[l], pb + 15);
        ln(Y, pb + 10, pb + 11, X, Xb);
    }
    ln(X, 44, 45, ENC, ENCb);

    // 5) decoder embedding
    embextra_kernel<<<(MTOT * DM) / 256, 256, 0, stream>>>(x_mark_dec, P(10), P(11), P(9), Y);
    gemm(xdnb, WembD, nullptr, Y, X, Xb, MTOT, DM, EINC, 0);

    // 6) decoder: self-attn (tau only)
    gemm(Xb, Sq, P(47), nullptr, nullptr, qb, MTOT, DM, DM, 0);
    gemm(Xb, Sk, P(49), nullptr, nullptr, kb, MTOT, DM, DM, 0);
    gemm(Xb, Sv, P(51), nullptr, Y, nullptr, MTOT, DM, DM, 0);
    vt_kernel<<<(MTOT * DM) / 256, 256, 0, stream>>>(Y, vt);
    attn(qb, kb, vt, nullptr, Zb);
    gemm(Zb, So, P(53), X, Y, nullptr, MTOT, DM, DM, 0);
    ln(Y, 62, 63, X, Xb);

    //    cross-attn (tau + delta, K/V from encoder)
    gemm(Xb,   Cq, P(55), nullptr, nullptr, qb, MTOT, DM, DM, 0);
    gemm(ENCb, Ck, P(57), nullptr, nullptr, kb, MTOT, DM, DM, 0);
    gemm(ENCb, Cv, P(59), nullptr, Y, nullptr, MTOT, DM, DM, 0);
    vt_kernel<<<(MTOT * DM) / 256, 256, 0, stream>>>(Y, vt);
    attn(qb, kb, vt, delB, Zb);
    gemm(Zb, Co, P(61), X, Y, nullptr, MTOT, DM, DM, 0);
    ln(Y, 64, 65, Z, Zb);

    //    FFN + final norms
    ffn(Zb, Z, Y, DW1, 69, DW2, 71);
    ln(Y, 66, 67, X, Xb);
    ln(X, 72, 73, nullptr, qb);     // dec_norm -> bf16 only (reuse qb)

    // 7) projection and de-normalization
    gemm(qb, Wproj, P(75), nullptr, Z, nullptr, MTOT, 128, DM, 0);
    final_kernel<<<(BB * 512 * 128) / 256, 256, 0, stream>>>(Z, stdb, meanb, (float*)d_out);
}
